// MultiScaleTransformerDecoderLayer_18846316494954
// MI455X (gfx1250) — compile-verified, matrix operand order checked by reading, not run on hardware
//
#include <hip/hip_runtime.h>
#include <hip/hip_bf16.h>

// ---------------- constants (match reference) ----------------
#define D_MODEL 256
#define NHEAD   8
#define HD      32            // self-attn head dim (256/8)
#define DFF     2048
#define NQ      900
#define BS      4
#define NLVL    4
#define HW_TOT  13294
#define M_Q     (NQ * BS)     // 3600
#define M_MEM   (HW_TOT * BS) // 53176
#define QTILES  ((NQ + 15) / 16) // 57

typedef unsigned short u16;
typedef __attribute__((ext_vector_type(16))) unsigned short u16x16;
typedef __attribute__((ext_vector_type(16))) __bf16 v16bf;
typedef __attribute__((ext_vector_type(8)))  float  v8f;

// ---------------- bf16 helpers ----------------
// Let the backend legalize fptrunc f32->bf16 (RNE); on CDNA5 this can use the
// native convert (and pack adjacent converts) instead of 3-op software RNE.
__device__ __forceinline__ u16 f2bf(float f) {
    return __builtin_bit_cast(u16, (__bf16)f);
}

__device__ __forceinline__ v8f zerov() {
    v8f z;
#pragma unroll
    for (int i = 0; i < 8; ++i) z[i] = 0.f;
    return z;
}

__device__ __forceinline__ v8f wmma_bf16(u16x16 a, u16x16 b, v8f c) {
    return __builtin_amdgcn_wmma_f32_16x16x32_bf16(
        false, __builtin_bit_cast(v16bf, a),
        false, __builtin_bit_cast(v16bf, b),
        (short)0, c, false, false);
}

// ISA 16-bit operand layout helper: per-lane K index for VGPR slot v.
__device__ __forceinline__ int frag_k(int v, int lane) {
    return ((v & 4) << 2) + (((lane >> 4) & 1) << 3) + ((v & 3) << 1);
}

// Fragment whose per-lane row pointer is precomputed (row = lane&15 semantics),
// reading contiguous (k, k+1) pairs. Works for A-frags (row=m) and B-frags (row=n).
__device__ __forceinline__ u16x16 frag_row_f32(const float* rowp, float scale) {
    int lane = threadIdx.x & 31;
    u16x16 r;
#pragma unroll
    for (int v = 0; v < 8; ++v) {
        int k = frag_k(v, lane);
        float a = rowp[k], b = rowp[k + 1];
        r[2 * v]     = f2bf(a * scale);
        r[2 * v + 1] = f2bf(b * scale);
    }
    return r;
}

__device__ __forceinline__ u16x16 frag_row_u16(const u16* rowp) {
    int lane = threadIdx.x & 31;
    u16x16 r;
#pragma unroll
    for (int v = 0; v < 8; ++v) {
        int k = frag_k(v, lane);
        r[2 * v]     = rowp[k];
        r[2 * v + 1] = rowp[k + 1];
    }
    return r;
}

// Generic B fragment: element (k, n) via functor (k varies per element).
template <typename F>
__device__ __forceinline__ u16x16 make_bfrag(F f) {
    int lane = threadIdx.x & 31;
    int n = lane & 15;
    u16x16 r;
#pragma unroll
    for (int v = 0; v < 8; ++v) {
        int k = frag_k(v, lane);
        r[2 * v]     = f(k,     n);
        r[2 * v + 1] = f(k + 1, n);
    }
    return r;
}

// ---------------- generic WMMA GEMM: C = A[MxK] * B[KxN] (+bias)(+prev)(relu) ----------------
// 128 threads = 4 waves; 64x64 block tile; each wave does a 32x32 quadrant.
// Requirements (all met by our launches): K % 32 == 0, N % 64 == 0. M arbitrary.
template <bool ADD_PREV, bool RELU, bool OUT_BF16>
__global__ __launch_bounds__(128) void gemm_kernel(
    const float* __restrict__ A, const float* __restrict__ B,
    const float* __restrict__ bias, void* __restrict__ Cout,
    int M, int N, int K)
{
    __shared__ u16 As[64][40];   // 64x32 (+pad)
    __shared__ u16 Bs[32][72];   // 32x64 (+pad)
    int tid  = threadIdx.x;
    int wave = tid >> 5;
    int lane = tid & 31;
    int m0 = blockIdx.y * 64;
    int n0 = blockIdx.x * 64;
    int wm = (wave >> 1) * 32;
    int wn = (wave & 1)  * 32;

    v8f acc00 = zerov(), acc01 = zerov(), acc10 = zerov(), acc11 = zerov();

    for (int k0 = 0; k0 < K; k0 += 32) {
        // A tile: 64x32 floats, float4 loads, clamped rows (dup rows are store-guarded away)
        for (int i = tid * 4; i < 64 * 32; i += 128 * 4) {
            int r = i >> 5, c = i & 31;
            int gm = m0 + r; gm = gm < M ? gm : M - 1;
            float4 t = *(const float4*)(A + (size_t)gm * K + (k0 + c));
            As[r][c]     = f2bf(t.x);
            As[r][c + 1] = f2bf(t.y);
            As[r][c + 2] = f2bf(t.z);
            As[r][c + 3] = f2bf(t.w);
        }
        // B tile: 32x64 floats, float4 loads, no guards needed (N%64==0, K%32==0)
        for (int i = tid * 4; i < 32 * 64; i += 128 * 4) {
            int r = i >> 6, c = i & 63;
            float4 t = *(const float4*)(B + (size_t)(k0 + r) * N + (n0 + c));
            Bs[r][c]     = f2bf(t.x);
            Bs[r][c + 1] = f2bf(t.y);
            Bs[r][c + 2] = f2bf(t.z);
            Bs[r][c + 3] = f2bf(t.w);
        }
        __syncthreads();

        int m = lane & 15;
        u16x16 a0 = frag_row_u16(&As[wm + m][0]);
        u16x16 a1 = frag_row_u16(&As[wm + 16 + m][0]);
        u16x16 b0 = make_bfrag([&](int k, int n) { return Bs[k][wn + n]; });
        u16x16 b1 = make_bfrag([&](int k, int n) { return Bs[k][wn + 16 + n]; });

        acc00 = wmma_bf16(a0, b0, acc00);
        acc01 = wmma_bf16(a0, b1, acc01);
        acc10 = wmma_bf16(a1, b0, acc10);
        acc11 = wmma_bf16(a1, b1, acc11);
        __syncthreads();
    }

    int n  = lane & 15;
    int mb = (lane >> 4) << 3;
#pragma unroll
    for (int r = 0; r < 8; ++r) {
        float vals[4] = { acc00[r], acc01[r], acc10[r], acc11[r] };
#pragma unroll
        for (int q = 0; q < 4; ++q) {
            int gm = m0 + wm + ((q >> 1) * 16) + mb + r;
            int gn = n0 + wn + ((q & 1) * 16) + n;
            if (gm < M) {
                float val = vals[q] + bias[gn];
                if (ADD_PREV) val += ((const float*)Cout)[(size_t)gm * N + gn];
                if (RELU)     val = fmaxf(val, 0.f);
                if (OUT_BF16) ((u16*)Cout)[(size_t)gm * N + gn] = f2bf(val);
                else          ((float*)Cout)[(size_t)gm * N + gn] = val;
            }
        }
    }
}

// ---------------- level weights: softmax(q @ lvl_w + lvl_b) ----------------
__global__ __launch_bounds__(32) void lvl_weight_kernel(
    const float* __restrict__ q, const float* __restrict__ lw,
    const float* __restrict__ lb, float* __restrict__ outw)
{
    int row  = blockIdx.x;
    int lane = threadIdx.x;
    float s[NLVL];
#pragma unroll
    for (int l = 0; l < NLVL; ++l) {
        float acc = 0.f;
        for (int d = lane; d < D_MODEL; d += 32)
            acc += q[(size_t)row * D_MODEL + d] * lw[d * NLVL + l];
#pragma unroll
        for (int off = 16; off > 0; off >>= 1)
            acc += __shfl_xor(acc, off, 32);
        s[l] = acc + lb[l];
    }
    if (lane == 0) {
        float mx = s[0];
#pragma unroll
        for (int l = 1; l < NLVL; ++l) mx = fmaxf(mx, s[l]);
        float sum = 0.f;
#pragma unroll
        for (int l = 0; l < NLVL; ++l) { s[l] = __expf(s[l] - mx); sum += s[l]; }
        float inv = 1.f / sum;
#pragma unroll
        for (int l = 0; l < NLVL; ++l) outw[(size_t)row * NLVL + l] = s[l] * inv;
    }
}

// ---------------- self-attention: one wave per (qtile, b, h) ----------------
__global__ __launch_bounds__(32) void self_attn_kernel(
    const float* __restrict__ qbuf, const float* __restrict__ kbuf,
    const float* __restrict__ vbuf, float* __restrict__ obuf)
{
    __shared__ float Lg[16][936];    // logits / probabilities (928 used)
    __shared__ float rowsum[16];
    int q0 = blockIdx.x * 16;
    int b  = blockIdx.y;
    int h  = blockIdx.z;
    int lane = threadIdx.x;
    int m = lane & 15;
    const float scale = 0.17677669529663687f;  // 1/sqrt(32)

    // Q fragment (clamped row; tail rows are store-guarded)
    {
        int qi = q0 + m; qi = qi < NQ ? qi : NQ - 1;
        const float* qrow = qbuf + ((size_t)(qi * BS + b)) * D_MODEL + h * HD;
        u16x16 qa = frag_row_f32(qrow, scale);

        for (int kt = 0; kt < QTILES; ++kt) {
            int kj = kt * 16 + m; kj = kj < NQ ? kj : NQ - 1;
            const float* krow = kbuf + ((size_t)(kj * BS + b)) * D_MODEL + h * HD;
            u16x16 bk = frag_row_f32(krow, 1.f);
            v8f c = zerov();
            c = wmma_bf16(qa, bk, c);
            int mb = (lane >> 4) << 3;
#pragma unroll
            for (int r = 0; r < 8; ++r) Lg[mb + r][kt * 16 + m] = c[r];
        }
    }
    __syncthreads();

    // softmax: 2 lanes per row (halves of [0,928)), combine via shfl
    {
        int half = lane >> 4;
        int jb = half * 464, je = jb + 464;
        int jeN = je < NQ ? je : NQ;
        float mx = -1e30f;
        for (int j = jb; j < jeN; ++j) mx = fmaxf(mx, Lg[m][j]);
        mx = fmaxf(mx, __shfl_xor(mx, 16, 32));
        float s = 0.f;
        for (int j = jb; j < jeN; ++j) {
            float p = __expf(Lg[m][j] - mx);
            Lg[m][j] = p; s += p;
        }
        for (int j = jeN; j < je; ++j) Lg[m][j] = 0.f;   // zero pad cols 900..927
        s += __shfl_xor(s, 16, 32);
        if (half == 0) rowsum[m] = s;
    }
    __syncthreads();

    v8f acc0 = zerov(), acc1 = zerov();
    for (int ks = 0; ks < 29; ++ks) {       // 29*32 = 928 keys (padded with P=0)
        u16x16 pa = frag_row_f32(&Lg[m][ks * 32], 1.f);
        u16x16 b0 = make_bfrag([&](int k, int n) {
            int kj = ks * 32 + k; kj = kj < NQ ? kj : NQ - 1;
            return f2bf(vbuf[((size_t)(kj * BS + b)) * D_MODEL + h * HD + n]);
        });
        acc0 = wmma_bf16(pa, b0, acc0);
        u16x16 b1 = make_bfrag([&](int k, int n) {
            int kj = ks * 32 + k; kj = kj < NQ ? kj : NQ - 1;
            return f2bf(vbuf[((size_t)(kj * BS + b)) * D_MODEL + h * HD + 16 + n]);
        });
        acc1 = wmma_bf16(pa, b1, acc1);
    }

    int mb = (lane >> 4) << 3;
#pragma unroll
    for (int r = 0; r < 8; ++r) {
        int mm = mb + r, qi = q0 + mm;
        if (qi < NQ) {
            float inv = 1.f / rowsum[mm];
            size_t base = ((size_t)(qi * BS + b)) * D_MODEL + h * HD;
            obuf[base + m]      = acc0[r] * inv;
            obuf[base + 16 + m] = acc1[r] * inv;
        }
    }
}

// ---------------- cross-attention (flash, per level): one wave per (qtile,b,h,lvl) ----------------
__global__ __launch_bounds__(32) void cross_attn_kernel(
    const float* __restrict__ qcbuf, const float* __restrict__ qsbuf,
    const u16* __restrict__ kcb, const u16* __restrict__ kpb,
    const u16* __restrict__ vvb, const unsigned char* __restrict__ mask,
    float* __restrict__ part)
{
    const int starts[NLVL] = { 0, 10000, 12500, 13125 };
    const int sizes[NLVL]  = { 10000, 2500, 625, 169 };
    int q0  = blockIdx.x * 16;
    int b   = blockIdx.y;
    int h   = blockIdx.z & 7;
    int lvl = blockIdx.z >> 3;
    int s0 = starts[lvl], Lk = sizes[lvl];
    int lane = threadIdx.x;
    int m = lane & 15;
    const float scale = 0.125f;   // 1/sqrt(64)

    __shared__ float Sg[16][33];
    __shared__ u16   Pg[16][40];
    __shared__ float alphaS[16], sumS[16];

    int qi = q0 + m; qi = qi < NQ ? qi : NQ - 1;
    u16x16 aqc = frag_row_f32(qcbuf + ((size_t)(qi * BS + b)) * D_MODEL + h * 32, scale);
    u16x16 aqs = frag_row_f32(qsbuf + ((size_t)(qi * BS + b)) * D_MODEL + h * 32, scale);

    float Mrow = -1e30f, Lrow = 0.f;      // lanes 0..15 own row state
    v8f acc0 = zerov(), acc1 = zerov();
    int nch = (Lk + 31) >> 5;

    for (int ch = 0; ch < nch; ++ch) {
        int kb = ch << 5;
#pragma unroll
        for (int t = 0; t < 2; ++t) {
            int kj = kb + t * 16 + m;
            int kjc = kj < Lk ? kj : Lk - 1;
            v8f c = zerov();
            u16x16 bkc = frag_row_u16(kcb + ((size_t)((s0 + kjc) * BS + b)) * D_MODEL + h * 32);
            c = wmma_bf16(aqc, bkc, c);
            u16x16 bkp = frag_row_u16(kpb + ((size_t)((s0 + kjc) * BS + b)) * D_MODEL + h * 32);
            c = wmma_bf16(aqs, bkp, c);
            int mb = (lane >> 4) << 3;
            bool valid = (kj < Lk) & (mask[(size_t)b * HW_TOT + s0 + kjc] == 0);
#pragma unroll
            for (int r = 0; r < 8; ++r) Sg[mb + r][t * 16 + m] = valid ? c[r] : -1e30f;
        }
        __syncthreads();

        if (lane < 16) {
            float cmax = -1e30f;
#pragma unroll
            for (int j = 0; j < 32; ++j) cmax = fmaxf(cmax, Sg[m][j]);
            float Mn = fmaxf(Mrow, cmax);
            float a  = __expf(Mrow - Mn);
            float ss = 0.f;
#pragma unroll
            for (int j = 0; j < 32; ++j) {
                float p = __expf(Sg[m][j] - Mn);
                Pg[m][j] = f2bf(p);
                ss += p;
            }
            Lrow = Lrow * a + ss;
            Mrow = Mn;
            alphaS[m] = a;
        }
        __syncthreads();

        {
            int mb = (lane >> 4) << 3;
#pragma unroll
            for (int r = 0; r < 8; ++r) {
                float a = alphaS[mb + r];
                acc0[r] *= a; acc1[r] *= a;
            }
        }

        u16x16 pa = frag_row_u16(&Pg[m][0]);
        u16x16 b0 = make_bfrag([&](int k, int n) {
            int kj = kb + k; kj = kj < Lk ? kj : Lk - 1;
            return vvb[((size_t)((s0 + kj) * BS + b)) * D_MODEL + h * 32 + n];
        });
        acc0 = wmma_bf16(pa, b0, acc0);
        u16x16 b1 = make_bfrag([&](int k, int n) {
            int kj = kb + k; kj = kj < Lk ? kj : Lk - 1;
            return vvb[((size_t)((s0 + kj) * BS + b)) * D_MODEL + h * 32 + 16 + n];
        });
        acc1 = wmma_bf16(pa, b1, acc1);
        __syncthreads();
    }

    if (lane < 16) sumS[m] = Lrow;
    __syncthreads();

    int mb = (lane >> 4) << 3;
    float* dst = part + (size_t)lvl * M_Q * D_MODEL;
#pragma unroll
    for (int r = 0; r < 8; ++r) {
        int mm = mb + r, qo = q0 + mm;
        if (qo < NQ) {
            float inv = 1.f / sumS[mm];
            size_t base = ((size_t)(qo * BS + b)) * D_MODEL + h * 32;
            dst[base + m]      = acc0[r] * inv;
            dst[base + 16 + m] = acc1[r] * inv;
        }
    }
}

// ---------------- weighted combine over levels ----------------
__global__ __launch_bounds__(256) void combine_kernel(
    const float* __restrict__ part, const float* __restrict__ w,
    float* __restrict__ out)
{
    int i = blockIdx.x * 256 + threadIdx.x;
    if (i >= M_Q * D_MODEL) return;
    int row = i >> 8;
    float acc = 0.f;
#pragma unroll
    for (int l = 0; l < NLVL; ++l)
        acc += w[(size_t)row * NLVL + l] * part[(size_t)l * M_Q * D_MODEL + i];
    out[i] = acc;
}

// ---------------- fused residual add + LayerNorm (256 threads per row) ----------------
__global__ __launch_bounds__(256) void add_ln_kernel(
    const float* __restrict__ x, const float* __restrict__ y,
    const float* __restrict__ g, const float* __restrict__ bta,
    float* __restrict__ out)
{
    __shared__ float red[256];
    int row = blockIdx.x, t = threadIdx.x;
    float v = x[(size_t)row * D_MODEL + t] + y[(size_t)row * D_MODEL + t];
    red[t] = v; __syncthreads();
    for (int off = 128; off > 0; off >>= 1) { if (t < off) red[t] += red[t + off]; __syncthreads(); }
    float mu = red[0] * (1.f / D_MODEL);
    __syncthreads();
    float d = v - mu;
    red[t] = d * d; __syncthreads();
    for (int off = 128; off > 0; off >>= 1) { if (t < off) red[t] += red[t + off]; __syncthreads(); }
    float var = red[0] * (1.f / D_MODEL);
    float r = rsqrtf(var + 1e-5f);
    out[(size_t)row * D_MODEL + t] = d * r * g[t] + bta[t];
}

// ---------------- host launch ----------------
extern "C" void kernel_launch(void* const* d_in, const int* in_sizes, int n_in,
                              void* d_out, int out_size, void* d_ws, size_t ws_size,
                              hipStream_t stream)
{
    (void)in_sizes; (void)n_in; (void)out_size; (void)ws_size;
    const float* tgt       = (const float*)d_in[0];
    const float* memory    = (const float*)d_in[1];
    const float* pos       = (const float*)d_in[2];
    const float* query_pos = (const float*)d_in[3];
    const float* qsine     = (const float*)d_in[4];
    const unsigned char* mask = (const unsigned char*)d_in[5];
    // d_in[6] = level_start_index (unused; static)
    auto P = [&](int i) { return (const float*)d_in[i]; };
    const float *sa_qc_w = P(7),  *sa_qc_b = P(8);
    const float *sa_qp_w = P(9),  *sa_qp_b = P(10);
    const float *sa_kc_w = P(11), *sa_kc_b = P(12);
    const float *sa_kp_w = P(13), *sa_kp_b = P(14);
    const float *sa_v_w  = P(15), *sa_v_b  = P(16);
    const float *sa_o_w  = P(17), *sa_o_b  = P(18);
    const float *ca_qc_w = P(19), *ca_qc_b = P(20);
    const float *ca_kc_w = P(21), *ca_kc_b = P(22);
    const float *ca_v_w  = P(23), *ca_v_b  = P(24);
    const float *ca_kp_w = P(25), *ca_kp_b = P(26);
    const float *ca_qs_w = P(27), *ca_qs_b = P(28);
    const float *ca_o_w  = P(29), *ca_o_b  = P(30);
    const float *lvl_w   = P(31), *lvl_b   = P(32);
    const float *lin1_w  = P(33), *lin1_b  = P(34);
    const float *lin2_w  = P(35), *lin2_b  = P(36);
    const float *n1g = P(37), *n1b = P(38);
    const float *n2g = P(39), *n2b = P(40);
    const float *n3g = P(41), *n3b = P(42);

    char* wsb = (char*)d_ws;
    size_t off = 0;
    auto allocF = [&](size_t n) { float* p = (float*)(wsb + off); off += n * sizeof(float); return p; };
    auto allocH = [&](size_t n) { u16*   p = (u16*)(wsb + off);   off += ((n * sizeof(u16) + 3) & ~(size_t)3); return p; };

    const size_t RQ = (size_t)M_Q * D_MODEL;
    float* q_sa    = allocF(RQ);
    float* k_sa    = allocF(RQ);
    float* v_sa    = allocF(RQ);
    float* sa_att  = allocF(RQ);
    float* sa_proj = allocF(RQ);
    float* tgt1    = allocF(RQ);
    float* qc      = allocF(RQ);
    float* qse     = allocF(RQ);
    float* levw    = allocF((size_t)M_Q * NLVL);
    float* part    = allocF((size_t)NLVL * RQ);
    float* comb    = allocF(RQ);
    float* ca_proj = allocF(RQ);
    float* tgtB    = allocF(RQ);
    float* ffn1    = allocF((size_t)M_Q * DFF);
    float* ffn2    = allocF(RQ);
    u16* kc_bf = allocH((size_t)M_MEM * D_MODEL);
    u16* kp_bf = allocH((size_t)M_MEM * D_MODEL);
    u16* v_bf  = allocH((size_t)M_MEM * D_MODEL);

    dim3 gQ((D_MODEL + 63) / 64, (M_Q + 63) / 64);     // (4, 57)
    dim3 gM((D_MODEL + 63) / 64, (M_MEM + 63) / 64);   // (4, 831)
    dim3 gF1((DFF + 63) / 64, (M_Q + 63) / 64);        // (32, 57)

    // ---- self-attention projections ----
    gemm_kernel<false, false, false><<<gQ, 128, 0, stream>>>(tgt,       sa_qc_w, sa_qc_b, q_sa, M_Q, D_MODEL, D_MODEL);
    gemm_kernel<true,  false, false><<<gQ, 128, 0, stream>>>(query_pos, sa_qp_w, sa_qp_b, q_sa, M_Q, D_MODEL, D_MODEL);
    gemm_kernel<false, false, false><<<gQ, 128, 0, stream>>>(tgt,       sa_kc_w, sa_kc_b, k_sa, M_Q, D_MODEL, D_MODEL);
    gemm_kernel<true,  false, false><<<gQ, 128, 0, stream>>>(query_pos, sa_kp_w, sa_kp_b, k_sa, M_Q, D_MODEL, D_MODEL);
    gemm_kernel<false, false, false><<<gQ, 128, 0, stream>>>(tgt,       sa_v_w,  sa_v_b,  v_sa, M_Q, D_MODEL, D_MODEL);

    // ---- level weights from pre-norm self-attn query ----
    lvl_weight_kernel<<<M_Q, 32, 0, stream>>>(q_sa, lvl_w, lvl_b, levw);

    // ---- self-attention + out-proj + norm1 ----
    self_attn_kernel<<<dim3(QTILES, BS, NHEAD), 32, 0, stream>>>(q_sa, k_sa, v_sa, sa_att);
    gemm_kernel<false, false, false><<<gQ, 128, 0, stream>>>(sa_att, sa_o_w, sa_o_b, sa_proj, M_Q, D_MODEL, D_MODEL);
    add_ln_kernel<<<M_Q, 256, 0, stream>>>(tgt, sa_proj, n1g, n1b, tgt1);

    // ---- cross-attention projections ----
    gemm_kernel<false, false, false><<<gQ, 128, 0, stream>>>(tgt1,  ca_qc_w, ca_qc_b, qc,  M_Q, D_MODEL, D_MODEL);
    gemm_kernel<false, false, false><<<gQ, 128, 0, stream>>>(qsine, ca_qs_w, ca_qs_b, qse, M_Q, D_MODEL, D_MODEL);
    gemm_kernel<false, false, true ><<<gM, 128, 0, stream>>>(memory, ca_kc_w, ca_kc_b, kc_bf, M_MEM, D_MODEL, D_MODEL);
    gemm_kernel<false, false, true ><<<gM, 128, 0, stream>>>(pos,    ca_kp_w, ca_kp_b, kp_bf, M_MEM, D_MODEL, D_MODEL);
    gemm_kernel<false, false, true ><<<gM, 128, 0, stream>>>(memory, ca_v_w,  ca_v_b,  v_bf,  M_MEM, D_MODEL, D_MODEL);

    // ---- cross-attention per level, weighted combine, out-proj, norm2 ----
    cross_attn_kernel<<<dim3(QTILES, BS, NHEAD * NLVL), 32, 0, stream>>>(qc, qse, kc_bf, kp_bf, v_bf, mask, part);
    combine_kernel<<<(M_Q * D_MODEL + 255) / 256, 256, 0, stream>>>(part, levw, comb);
    gemm_kernel<false, false, false><<<gQ, 128, 0, stream>>>(comb, ca_o_w, ca_o_b, ca_proj, M_Q, D_MODEL, D_MODEL);
    add_ln_kernel<<<M_Q, 256, 0, stream>>>(tgt1, ca_proj, n2g, n2b, tgtB);

    // ---- FFN + norm3 -> output ----
    gemm_kernel<false, true,  false><<<gF1, 128, 0, stream>>>(tgtB, lin1_w, lin1_b, ffn1, M_Q, DFF, D_MODEL);
    gemm_kernel<false, false, false><<<gQ, 128, 0, stream>>>(ffn1, lin2_w, lin2_b, ffn2, M_Q, D_MODEL, DFF);
    add_ln_kernel<<<M_Q, 256, 0, stream>>>(tgtB, ffn2, n3g, n3b, (float*)d_out);
}
